// ClusterBasedVSDGATRNN_53523882442790
// MI455X (gfx1250) — compile-verified, matrix operand order checked by reading, not run on hardware
//
#include <hip/hip_runtime.h>
#include <math.h>

// ---- problem dims (fixed by reference) ----
#define NN   4096   // nodes
#define CC   64     // clusters
#define DD   128    // hidden
#define HH   4      // heads
#define CDIM 256    // combined dim = 2*DD
#define DQ   32     // per-head q/k dim

typedef __attribute__((ext_vector_type(16))) __bf16 v16bf;
typedef __attribute__((ext_vector_type(8)))  float  v8f;
typedef int v4i_ __attribute__((vector_size(16)));   // matches builtin param type

// ---- gfx1250 async-to-LDS capability probe ----
#if defined(__has_builtin)
#if __has_builtin(__builtin_amdgcn_global_load_async_to_lds_b128)
#define HAVE_ASYNC_LDS 1
#warning "CDNA5: using global_load_async_to_lds_b128 staging path"
#else
#define HAVE_ASYNC_LDS 0
#warning "CDNA5: async-to-LDS builtin NOT found; sync LDS staging fallback"
#endif
#else
#define HAVE_ASYNC_LDS 0
#endif

#if HAVE_ASYNC_LDS
#define ASYNC_GPTR(p) ((__attribute__((address_space(1))) v4i_*)(p))
#define ASYNC_LPTR(p) ((__attribute__((address_space(3))) v4i_*)(p))
#endif

// ---------------- bf16 helpers (RNE) ----------------
__device__ __forceinline__ unsigned short f2bf(float f) {
  union { float f; unsigned u; } v; v.f = f;
  unsigned r = v.u + 0x7FFFu + ((v.u >> 16) & 1u);
  return (unsigned short)(r >> 16);
}

// ---------------- WMMA fragment loaders (wave32, 16x16x32 bf16) ----------------
// A matrix 16x32: lanes 0-15 -> M=lane, lanes 16-31 -> M=lane-16 (K offset +8)
// element pair u[2v],u[2v+1]: K = (v<4 ? 2v : 16+2(v-4)) + half*8 (+0/+1)
__device__ __forceinline__ v16bf load_a(const unsigned short* t, int ld) {
  int lane = threadIdx.x & 31; int m = lane & 15; int hi = lane >> 4;
  union { v16bf v; unsigned w[8]; } r;
#pragma unroll
  for (int vv = 0; vv < 8; ++vv) {
    int k = ((vv < 4) ? 2 * vv : 16 + 2 * (vv - 4)) + hi * 8;
    r.w[vv] = *(const unsigned*)(t + m * ld + k);     // 2 consecutive bf16
  }
  return r.v;
}
// B (K x 16) given transposed storage t[n][k] (weight rows = output cols)
__device__ __forceinline__ v16bf load_bt(const unsigned short* t, int ld) {
  int lane = threadIdx.x & 31; int n = lane & 15; int hi = lane >> 4;
  union { v16bf v; unsigned w[8]; } r;
#pragma unroll
  for (int vv = 0; vv < 8; ++vv) {
    int k = ((vv < 4) ? 2 * vv : 16 + 2 * (vv - 4)) + hi * 8;
    r.w[vv] = *(const unsigned*)(t + n * ld + k);
  }
  return r.v;
}
// B (K x 16) natural storage t[k][n] (e.g. V tile) -- strided 16-bit loads
__device__ __forceinline__ v16bf load_bn(const unsigned short* t, int ld) {
  int lane = threadIdx.x & 31; int n = lane & 15; int hi = lane >> 4;
  union { v16bf v; unsigned short u[16]; } r;
#pragma unroll
  for (int vv = 0; vv < 8; ++vv) {
    int k = ((vv < 4) ? 2 * vv : 16 + 2 * (vv - 4)) + hi * 8;
    r.u[2 * vv]     = t[(k)     * ld + n];
    r.u[2 * vv + 1] = t[(k + 1) * ld + n];
  }
  return r.v;
}

// ---------------- elementwise / conversion kernels ----------------
__global__ void build_combined(const float* __restrict__ x, const float* __restrict__ h,
                               unsigned short* __restrict__ comb) {
  int tid = blockIdx.x * blockDim.x + threadIdx.x;
  if (tid >= NN * CDIM) return;
  int n = tid >> 8, j = tid & 255;
  float v = (j < DD) ? x[n * DD + j] : h[n * DD + (j - DD)];
  comb[tid] = f2bf(v);
}
__global__ void cvt_bf16(const float* __restrict__ src, unsigned short* __restrict__ dst, int n) {
  int t = blockIdx.x * blockDim.x + threadIdx.x;
  if (t < n) dst[t] = f2bf(src[t]);
}
__global__ void zero_f32(float* p, int n) {
  int t = blockIdx.x * blockDim.x + threadIdx.x;
  if (t < n) p[t] = 0.f;
}

// ---------------- generic batched WMMA GEMM: C = A * B^T + bias ----------------
// A: [M,K] bf16 row-major; Bw: [Nc,K] bf16 row-major (weights); bias f32 [Nc]
// grid: (M/128, Nc/16, batch); block 256 = 8 waves, each wave does one 16x16 tile
__global__ __launch_bounds__(256)
void gemm_bf16(const unsigned short* __restrict__ A, long aBatch,
               const unsigned short* __restrict__ Bw, long bBatch,
               const float* __restrict__ bias, long biasBatch,
               float* Cf, unsigned short* Cb, long cBatch,
               int M, int Nc, int K, int doRelu) {
  int wave = threadIdx.x >> 5;
  int row0 = (blockIdx.x * 8 + wave) * 16;
  int col0 = blockIdx.y * 16;
  int z = blockIdx.z;
  if (row0 >= M) return;
  const unsigned short* Ab = A + (long)z * aBatch + (long)row0 * K;
  const unsigned short* Bb = Bw + (long)z * bBatch + (long)col0 * K;
  v8f acc = {0.f, 0.f, 0.f, 0.f, 0.f, 0.f, 0.f, 0.f};
  for (int k0 = 0; k0 < K; k0 += 32) {
    v16bf fa = load_a(Ab + k0, K);
    v16bf fb = load_bt(Bb + k0, K);
    acc = __builtin_amdgcn_wmma_f32_16x16x32_bf16(false, fa, false, fb,
                                                  (short)0, acc, false, false);
  }
  int lane = threadIdx.x & 31; int n = lane & 15; int hi = lane >> 4;
  float bsv = bias ? bias[(long)z * biasBatch + col0 + n] : 0.f;
#pragma unroll
  for (int r = 0; r < 8; ++r) {
    float v = acc[r] + bsv;
    if (doRelu) v = fmaxf(v, 0.f);
    long idx = (long)z * cBatch + (long)(row0 + r + 8 * hi) * Nc + col0 + n;
    if (Cf) Cf[idx] = v;
    if (Cb) Cb[idx] = f2bf(v);
  }
}

// ---------------- K/V tile staging (async-to-LDS when available) ----------------
// K tile: 32x32 bf16 (2 KB), V tile: 32x256 bf16 (16 KB); block = 256 threads.
__device__ __forceinline__ void stage_tiles(const unsigned short* __restrict__ kh,
                                            const unsigned short* __restrict__ vh,
                                            int j0, unsigned short* sK, unsigned short* sV) {
  int t = threadIdx.x;
#if HAVE_ASYNC_LDS
  if (t < 128) {                       // K: 128 x b128 = 2048 B
    int idx = t * 8;                   // ushort index
    int key = idx >> 5, dq = idx & 31;
    __builtin_amdgcn_global_load_async_to_lds_b128(
        ASYNC_GPTR(kh + (long)(j0 + key) * DQ + dq),
        ASYNC_LPTR(sK + idx), 0, 0);
  }
#pragma unroll
  for (int e = 0; e < 4; ++e) {        // V: 1024 x b128 = 16384 B
    int idx = (t * 4 + e) * 8;
    int key = idx >> 8, col = idx & 255;
    __builtin_amdgcn_global_load_async_to_lds_b128(
        ASYNC_GPTR(vh + (long)(j0 + key) * CDIM + col),
        ASYNC_LPTR(sV + idx), 0, 0);
  }
#else
#pragma unroll
  for (int e = 0; e < 4; ++e) {
    int idx = t * 4 + e;
    int key = idx >> 5, dq = idx & 31;
    sK[idx] = kh[(long)(j0 + key) * DQ + dq];
  }
  {
    int key = t >> 3, cb = (t & 7) * 32;
#pragma unroll
    for (int e = 0; e < 32; ++e)
      sV[key * CDIM + cb + e] = vh[(long)(j0 + key) * CDIM + cb + e];
  }
#endif
}

// ---------------- flash attention (per head), leaky_relu(0.2) + softmax ----------------
// grid (NN/128, HH), block 256. 8 waves share double-buffered LDS K/V tiles;
// each wave owns one 16-query tile. Online softmax, P re-fragmented through LDS.
__global__ __launch_bounds__(256)
void attn_kernel(const unsigned short* __restrict__ qg,
                 const unsigned short* __restrict__ kg,
                 const unsigned short* __restrict__ vg,
                 unsigned short* __restrict__ hp) {
  __shared__ __align__(16) unsigned short sK[2][32 * DQ];     // [buf][key][dq]
  __shared__ __align__(16) unsigned short sV[2][32 * CDIM];   // [buf][key][e]
  __shared__ __align__(16) unsigned short sP[8][16 * 32];     // per-wave P tile

  const int head = blockIdx.y;
  const int wave = threadIdx.x >> 5;
  const int lane = threadIdx.x & 31;
  const int row0 = (blockIdx.x * 8 + wave) * 16;
  const int n16 = lane & 15, hi = lane >> 4;

  const unsigned short* qh = qg + ((long)head * NN + row0) * DQ;
  const unsigned short* kh = kg + (long)head * NN * DQ;
  const unsigned short* vh = vg + (long)head * NN * CDIM;

  v16bf fq = load_a(qh, DQ);
  v8f zerov = {0.f, 0.f, 0.f, 0.f, 0.f, 0.f, 0.f, 0.f};
  v8f acc[16];
#pragma unroll
  for (int t = 0; t < 16; ++t) acc[t] = zerov;
  float mrun[8], lrun[8];
#pragma unroll
  for (int r = 0; r < 8; ++r) { mrun[r] = -INFINITY; lrun[r] = 0.f; }
  const float scale = 0.17677669529663687f;  // 1/sqrt(DQ)

  const int NIT = NN / 32;
  stage_tiles(kh, vh, 0, sK[0], sV[0]);       // prologue: fill buffer 0

  for (int it = 0; it < NIT; ++it) {
    const int cur = it & 1;
#if HAVE_ASYNC_LDS
    asm volatile("s_wait_asynccnt 0" ::: "memory");   // my async tile writes done
#endif
    __syncthreads();                                  // whole tile visible; prev compute done
    if (it + 1 < NIT)                                 // overlap: fetch next tile into other buffer
      stage_tiles(kh, vh, (it + 1) * 32, sK[cur ^ 1], sV[cur ^ 1]);

    const unsigned short* cK = sK[cur];
    const unsigned short* cV = sV[cur];

    // scores: Q(16x32) x K^T(32x16) twice (two key sub-blocks)
    v16bf fb0 = load_bt(cK, DQ);
    v16bf fb1 = load_bt(cK + 16 * DQ, DQ);
    v8f s0 = __builtin_amdgcn_wmma_f32_16x16x32_bf16(false, fq, false, fb0,
                                                     (short)0, zerov, false, false);
    v8f s1 = __builtin_amdgcn_wmma_f32_16x16x32_bf16(false, fq, false, fb1,
                                                     (short)0, zerov, false, false);
    unsigned short* pw = sP[wave];
#pragma unroll
    for (int r = 0; r < 8; ++r) {
      float a = s0[r] * scale; a = (a > 0.f) ? a : 0.2f * a;   // leaky_relu
      float b = s1[r] * scale; b = (b > 0.f) ? b : 0.2f * b;
      float mx = fmaxf(a, b);
#pragma unroll
      for (int off = 8; off >= 1; off >>= 1) mx = fmaxf(mx, __shfl_xor(mx, off, 32));
      float mnew = fmaxf(mrun[r], mx);
      float corr = __expf(mrun[r] - mnew);
      float pa = __expf(a - mnew), pb = __expf(b - mnew);
      float rs = pa + pb;
#pragma unroll
      for (int off = 8; off >= 1; off >>= 1) rs += __shfl_xor(rs, off, 32);
      lrun[r] = lrun[r] * corr + rs;
      mrun[r] = mnew;
#pragma unroll
      for (int t = 0; t < 16; ++t) acc[t][r] *= corr;
      int m = r + 8 * hi;
      pw[m * 32 + n16]      = f2bf(pa);
      pw[m * 32 + 16 + n16] = f2bf(pb);
    }
    asm volatile("s_wait_dscnt 0" ::: "memory");  // P store -> P reload (same wave)
    v16bf fp = load_a(pw, 32);
#pragma unroll
    for (int t = 0; t < 16; ++t) {
      v16bf fv = load_bn(cV + t * 16, CDIM);
      acc[t] = __builtin_amdgcn_wmma_f32_16x16x32_bf16(false, fp, false, fv,
                                                       (short)0, acc[t], false, false);
    }
  }
  // epilogue: normalize, write hp[n][head*CDIM + e] as bf16
#pragma unroll
  for (int r = 0; r < 8; ++r) {
    float inv = 1.f / lrun[r];
    int m = row0 + r + 8 * hi;
#pragma unroll
    for (int t = 0; t < 16; ++t)
      hp[(long)m * (HH * CDIM) + head * CDIM + t * 16 + n16] = f2bf(acc[t][r] * inv);
  }
}

// ---------------- segment mean + GRU + cluster->node ----------------
__global__ void seg_sum(const float* __restrict__ h, const int* __restrict__ lab,
                        float* sums, float* counts) {
  int n = blockIdx.x; int j = threadIdx.x;  // grid NN, block DD
  int c = lab[n];
  atomicAdd(&sums[c * DD + j], h[n * DD + j]);
  if (j == 0) atomicAdd(&counts[c], 1.f);
}

__global__ void gru_kernel(const float* __restrict__ sums, const float* __restrict__ counts,
                           const float* __restrict__ cluster_h,
                           const float* __restrict__ Wih, const float* __restrict__ bih,
                           const float* __restrict__ Whh, const float* __restrict__ bhh,
                           float* __restrict__ outC) {
  __shared__ float sa[DD], sh[DD];
  int c = blockIdx.x, j = threadIdx.x;  // grid CC, block DD
  float cnt = counts[c];
  sa[j] = sums[c * DD + j] / fmaxf(cnt, 1.f);
  sh[j] = cluster_h[c * DD + j];
  __syncthreads();
  float ir = bih[j], iz = bih[DD + j], in = bih[2 * DD + j];
  float hr = bhh[j], hz = bhh[DD + j], hn = bhh[2 * DD + j];
  for (int d = 0; d < DD; ++d) {
    float a = sa[d], hh = sh[d];
    ir += a * Wih[j * DD + d];
    iz += a * Wih[(DD + j) * DD + d];
    in += a * Wih[(2 * DD + j) * DD + d];
    hr += hh * Whh[j * DD + d];
    hz += hh * Whh[(DD + j) * DD + d];
    hn += hh * Whh[(2 * DD + j) * DD + d];
  }
  float r = 1.f / (1.f + __expf(-(ir + hr)));
  float z = 1.f / (1.f + __expf(-(iz + hz)));
  float nn = tanhf(in + r * hn);
  float newh = (1.f - z) * nn + z * sh[j];
  outC[c * DD + j] = (cnt > 0.f) ? newh : sh[j];
}

__global__ void c2n_kernel(const float* __restrict__ uch, const float* __restrict__ Wp,
                           const float* __restrict__ bp, float* __restrict__ c2n) {
  __shared__ float sc[DD];
  int c = blockIdx.x, j = threadIdx.x;  // grid CC, block DD
  sc[j] = uch[c * DD + j];
  __syncthreads();
  float a = bp[j];
  for (int d = 0; d < DD; ++d) a += sc[d] * Wp[j * DD + d];
  c2n[c * DD + j] = a;
}

__global__ void final_kernel(const float* __restrict__ tail, const float* __restrict__ h,
                             const float* __restrict__ c2n, const int* __restrict__ lab,
                             float* __restrict__ out) {
  int tid = blockIdx.x * blockDim.x + threadIdx.x;
  if (tid >= NN * DD) return;
  int n = tid >> 7, j = tid & 127;
  out[tid] = tail[tid] + h[tid] + c2n[lab[n] * DD + j];
}

// ---------------- launcher ----------------
extern "C" void kernel_launch(void* const* d_in, const int* in_sizes, int n_in,
                              void* d_out, int out_size, void* d_ws, size_t ws_size,
                              hipStream_t stream) {
  const float* x   = (const float*)d_in[0];
  const float* h   = (const float*)d_in[1];
  const float* clh = (const float*)d_in[2];
  const int*   lab = (const int*)d_in[4];
  const float* Wq  = (const float*)d_in[8];   const float* bq = (const float*)d_in[9];
  const float* Wk  = (const float*)d_in[10];  const float* bk = (const float*)d_in[11];
  const float* Wv  = (const float*)d_in[12];  const float* bv = (const float*)d_in[13];
  const float* W1  = (const float*)d_in[14];  const float* b1 = (const float*)d_in[15];
  const float* W2  = (const float*)d_in[16];  const float* b2 = (const float*)d_in[17];
  const float* Wih = (const float*)d_in[18];  const float* bih = (const float*)d_in[19];
  const float* Whh = (const float*)d_in[20];  const float* bhh = (const float*)d_in[21];
  const float* Wp  = (const float*)d_in[22];  const float* bp = (const float*)d_in[23];

  float* outH = (float*)d_out;            // updated_h [NN,DD]
  float* outC = outH + (long)NN * DD;     // updated_cluster_h [CC,DD]

  // workspace carve-up (bytes, 256-aligned)
  char* ws = (char*)d_ws;
  size_t off = 0;
  auto take = [&](size_t bytes) { char* p = ws + off; off = (off + bytes + 255) & ~(size_t)255; return p; };
  unsigned short* comb = (unsigned short*)take((size_t)NN * CDIM * 2);
  unsigned short* wqb  = (unsigned short*)take((size_t)HH * DQ * CDIM * 2);
  unsigned short* wkb  = (unsigned short*)take((size_t)HH * DQ * CDIM * 2);
  unsigned short* wvb  = (unsigned short*)take((size_t)HH * CDIM * CDIM * 2);
  unsigned short* w1b  = (unsigned short*)take((size_t)CDIM * HH * CDIM * 2);
  unsigned short* w2b  = (unsigned short*)take((size_t)CDIM * CDIM * 2);
  unsigned short* qb   = (unsigned short*)take((size_t)HH * NN * DQ * 2);
  unsigned short* kb   = (unsigned short*)take((size_t)HH * NN * DQ * 2);
  unsigned short* vb   = (unsigned short*)take((size_t)HH * NN * CDIM * 2);
  unsigned short* hpb  = (unsigned short*)take((size_t)NN * HH * CDIM * 2);
  unsigned short* h1b  = (unsigned short*)take((size_t)NN * CDIM * 2);
  float* tail  = (float*)take((size_t)NN * DD * 4);
  float* sums  = (float*)take((size_t)(CC * DD + CC) * 4);   // sums then counts
  float* cnts  = sums + CC * DD;
  float* c2n   = (float*)take((size_t)CC * DD * 4);
  (void)ws_size; (void)n_in; (void)in_sizes; (void)out_size;

  // 1) build bf16 combined + weights
  build_combined<<<(NN * CDIM + 255) / 256, 256, 0, stream>>>(x, h, comb);
  cvt_bf16<<<(HH * DQ * CDIM + 255) / 256, 256, 0, stream>>>(Wq, wqb, HH * DQ * CDIM);
  cvt_bf16<<<(HH * DQ * CDIM + 255) / 256, 256, 0, stream>>>(Wk, wkb, HH * DQ * CDIM);
  cvt_bf16<<<(HH * CDIM * CDIM + 255) / 256, 256, 0, stream>>>(Wv, wvb, HH * CDIM * CDIM);
  cvt_bf16<<<(CDIM * HH * CDIM + 255) / 256, 256, 0, stream>>>(W1, w1b, CDIM * HH * CDIM);
  cvt_bf16<<<(CDIM * CDIM + 255) / 256, 256, 0, stream>>>(W2, w2b, CDIM * CDIM);

  // 2) QKV projections (batched over heads via grid.z)
  gemm_bf16<<<dim3(NN / 128, DQ / 16, HH), 256, 0, stream>>>(
      comb, 0L, wqb, (long)DQ * CDIM, bq, (long)DQ,
      nullptr, qb, (long)NN * DQ, NN, DQ, CDIM, 0);
  gemm_bf16<<<dim3(NN / 128, DQ / 16, HH), 256, 0, stream>>>(
      comb, 0L, wkb, (long)DQ * CDIM, bk, (long)DQ,
      nullptr, kb, (long)NN * DQ, NN, DQ, CDIM, 0);
  gemm_bf16<<<dim3(NN / 128, CDIM / 16, HH), 256, 0, stream>>>(
      comb, 0L, wvb, (long)CDIM * CDIM, bv, (long)CDIM,
      nullptr, vb, (long)NN * CDIM, NN, CDIM, CDIM, 0);

  // 3) flash attention -> hp bf16 [NN, HH*CDIM]
  attn_kernel<<<dim3(NN / 128, HH), 256, 0, stream>>>(qb, kb, vb, hpb);

  // 4) MLP: h1 = relu(hp@W1^T+b1); tail = (h1@W2[128:]^T + b2[128:])  (only needed half)
  gemm_bf16<<<dim3(NN / 128, CDIM / 16, 1), 256, 0, stream>>>(
      hpb, 0L, w1b, 0L, b1, 0L, nullptr, h1b, 0L, NN, CDIM, HH * CDIM, 1);
  gemm_bf16<<<dim3(NN / 128, DD / 16, 1), 256, 0, stream>>>(
      h1b, 0L, w2b + (size_t)DD * CDIM, 0L, b2 + DD, 0L,
      tail, nullptr, 0L, NN, DD, CDIM, 0);

  // 5) segment mean -> GRU -> updated cluster h (output 2)
  zero_f32<<<(CC * DD + CC + 255) / 256, 256, 0, stream>>>(sums, CC * DD + CC);
  seg_sum<<<NN, DD, 0, stream>>>(h, lab, sums, cnts);
  gru_kernel<<<CC, DD, 0, stream>>>(sums, cnts, clh, Wih, bih, Whh, bhh, outC);

  // 6) cluster->node projection and final add (output 1)
  c2n_kernel<<<CC, DD, 0, stream>>>(outC, Wp, bp, c2n);
  final_kernel<<<(NN * DD + 255) / 256, 256, 0, stream>>>(tail, h, c2n, lab, outH);
}
